// SparseAttention_2619930051614
// MI455X (gfx1250) — matrix operand order repaired and checked by reading, not compiled, not run on hardware
//
#include <hip/hip_runtime.h>
#include <hip/hip_bf16.h>
#include <math.h>

// ---------------------------------------------------------------------------
// Sparse top-k attention for MI455X (gfx1250), wave32 + WMMA bf16.
//
// Shapes: B=4, S=2048, D=512, H=8, dh=64, top_k=10, scale=1/8.
// d_out = [context (4*2048*512 f32)] ++ [attn (4*8*2048*2048 f32)].
// The 537MB attn write dominates (roofline ~23us @ 23.3TB/s); matmuls use
// v_wmma_f32_16x16x32_bf16 so compute stays well under the memory floor.
// ---------------------------------------------------------------------------

typedef __attribute__((ext_vector_type(16))) __bf16 v16bf;
typedef __attribute__((ext_vector_type(8)))  float  v8f;
typedef __attribute__((ext_vector_type(4)))  unsigned int u32x4;
typedef __attribute__((ext_vector_type(4)))  float  f32x4;

union FragBF { v16bf v; u32x4 q[2]; };

#define BATCH 4
#define SEQ   2048
#define DMODEL 512
#define HEADS 8
#define DHEAD 64
#define TOPK  10
#define CTX_ELEMS (BATCH * SEQ * DMODEL)          // 4,194,304
#define NEG_BIG  (-3.0e38f)

// ------------------------------ f32 -> bf16 --------------------------------
__global__ __launch_bounds__(256) void f32_to_bf16_kernel(
    const float* __restrict__ in, __bf16* __restrict__ out, int n) {
  for (int i = blockIdx.x * blockDim.x + threadIdx.x; i < n;
       i += gridDim.x * blockDim.x) {
    out[i] = (__bf16)in[i];
  }
}

// -------------------------- QKV projection GEMM ----------------------------
// y[m,n] = sum_k x[m,k] * W[n,k] + b[n]   (torch Linear: x @ W.T + b)
// Wave computes a 16(M) x 64(N) tile; block = 8 waves = 128(M) x 64(N).
// Output scattered to head-major [B, H, S, dh] layout.
template <bool BF16OUT>
__global__ __launch_bounds__(256) void proj_gemm_kernel(
    const __bf16* __restrict__ xbf, const __bf16* __restrict__ wbf,
    const float* __restrict__ bias, void* __restrict__ outp) {
  const int lane = threadIdx.x & 31;
  const int w    = threadIdx.x >> 5;
  const int m0   = (blockIdx.x * 8 + w) * 16;   // 0..8176
  const int n0   = blockIdx.y * 64;             // 0..448

  v8f acc[4];
#pragma unroll
  for (int j = 0; j < 4; ++j) acc[j] = (v8f){0, 0, 0, 0, 0, 0, 0, 0};

  const int arow  = m0 + (lane & 15);
  const int abase = (lane >> 4) << 3;   // {0,8}
  const int bn    = lane & 15;
  const int bbase = (lane >> 4) << 3;   // {0,8}

  for (int k0 = 0; k0 < DMODEL; k0 += 32) {
    FragBF a;
    const __bf16* ap = xbf + (size_t)arow * DMODEL + k0 + abase;
    a.q[0] = *(const u32x4*)(ap);
    a.q[1] = *(const u32x4*)(ap + 16);
#pragma unroll
    for (int j = 0; j < 4; ++j) {
      FragBF b;
      const __bf16* bp =
          wbf + (size_t)(n0 + j * 16 + bn) * DMODEL + k0 + bbase;
      b.q[0] = *(const u32x4*)(bp);
      b.q[1] = *(const u32x4*)(bp + 16);
      acc[j] = __builtin_amdgcn_wmma_f32_16x16x32_bf16(
          false, a.v, false, b.v, (short)0, acc[j], false, false);
    }
  }

#pragma unroll
  for (int j = 0; j < 4; ++j) {
    const int n    = n0 + j * 16 + bn;
    const float bv = bias[n];
    const int head = n >> 6, dd = n & 63;
#pragma unroll
    for (int r = 0; r < 8; ++r) {
      const int m = m0 + r + ((lane >> 4) << 3);
      const int b = m >> 11, s = m & (SEQ - 1);
      const size_t off = (((size_t)(b * HEADS + head) * SEQ) + s) * DHEAD + dd;
      const float val = acc[j][r] + bv;
      if (BF16OUT)
        ((__bf16*)outp)[off] = (__bf16)val;
      else
        ((float*)outp)[off] = val;
    }
  }
}

// ------------------------- top-k helpers (wave32) --------------------------
__device__ __forceinline__ void topk_insert(float (&tv)[TOPK], int (&ti)[TOPK],
                                            float v, int idx) {
  if (v > tv[TOPK - 1] || (v == tv[TOPK - 1] && idx < ti[TOPK - 1])) {
    int p = TOPK - 1;
    while (p > 0 && (v > tv[p - 1] || (v == tv[p - 1] && idx < ti[p - 1]))) {
      tv[p] = tv[p - 1];
      ti[p] = ti[p - 1];
      --p;
    }
    tv[p] = v;
    ti[p] = idx;
  }
}

// Butterfly argmax across 32 lanes; tie-break by lower index (matches
// jax.lax.top_k stability). All lanes converge to the same (v, i).
__device__ __forceinline__ void wave_argmax(float& v, int& i) {
#pragma unroll
  for (int off = 16; off; off >>= 1) {
    float vo = __shfl_xor(v, off, 32);
    int   io = __shfl_xor(i, off, 32);
    if (vo > v || (vo == v && io < i)) { v = vo; i = io; }
  }
}

// ------------------- fused scores/top-k/softmax/output ---------------------
// Block = 256 threads (8 waves) handles one (b,h, 16-row q-tile).
// Scores streamed through LDS in 16x512 f32 chunks (32 KB).
__global__ __launch_bounds__(256) void sparse_attn_kernel(
    const __bf16* __restrict__ Qbf, const __bf16* __restrict__ Kbf,
    const float* __restrict__ Vf, float* __restrict__ out) {
  extern __shared__ float sc[];  // [16][512]
  const int lane = threadIdx.x & 31;
  const int w    = threadIdx.x >> 5;
  const int bh   = blockIdx.y;          // 0..31  (b*8 + head)
  const int q0   = blockIdx.x * 16;

  const __bf16* Qh = Qbf + (size_t)bh * SEQ * DHEAD;
  const __bf16* Kh = Kbf + (size_t)bh * SEQ * DHEAD;
  const float*  Vh = Vf  + (size_t)bh * SEQ * DHEAD;

  // Q tile 16x64 = two 16x32 A-fragments, held for the whole kernel.
  FragBF A0, A1;
  {
    const int arow  = q0 + (lane & 15);
    const int abase = (lane >> 4) << 3;
    const __bf16* ap = Qh + (size_t)arow * DHEAD + abase;
    A0.q[0] = *(const u32x4*)(ap);
    A0.q[1] = *(const u32x4*)(ap + 16);
    A1.q[0] = *(const u32x4*)(ap + 32);
    A1.q[1] = *(const u32x4*)(ap + 48);
  }

  // Per-lane running top-10 for the two rows this wave owns in phase 2.
  float lv[2][TOPK];
  int   li[2][TOPK];
#pragma unroll
  for (int r = 0; r < 2; ++r)
#pragma unroll
    for (int t = 0; t < TOPK; ++t) { lv[r][t] = NEG_BIG; li[r][t] = 0x7fffffff; }
  const int r0 = w * 2;

  for (int cb = 0; cb < SEQ; cb += 512) {
    // Phase 1: wave w computes score columns [cb + w*64, cb + w*64 + 64).
#pragma unroll
    for (int ct = 0; ct < 4; ++ct) {
      const int col0 = cb + w * 64 + ct * 16;
      FragBF B0, B1;
      const __bf16* kp =
          Kh + (size_t)(col0 + (lane & 15)) * DHEAD + ((lane >> 4) << 3);
      B0.q[0] = *(const u32x4*)(kp);
      B0.q[1] = *(const u32x4*)(kp + 16);
      B1.q[0] = *(const u32x4*)(kp + 32);
      B1.q[1] = *(const u32x4*)(kp + 48);
      v8f acc = (v8f){0, 0, 0, 0, 0, 0, 0, 0};
      acc = __builtin_amdgcn_wmma_f32_16x16x32_bf16(
          false, A0.v, false, B0.v, (short)0, acc, false, false);
      acc = __builtin_amdgcn_wmma_f32_16x16x32_bf16(
          false, A1.v, false, B1.v, (short)0, acc, false, false);
      const int lcol = (col0 - cb) + (lane & 15);
#pragma unroll
      for (int r = 0; r < 8; ++r) {
        const int row = r + ((lane >> 4) << 3);
        sc[row * 512 + lcol] = acc[r] * 0.125f;  // 1/sqrt(64)
      }
    }
    __syncthreads();

    // Phase 2: scan this chunk of rows r0, r0+1 into per-lane top-10.
#pragma unroll
    for (int rI = 0; rI < 2; ++rI) {
      const float* rowp = sc + (r0 + rI) * 512;
#pragma unroll
      for (int j = 0; j < 16; ++j) {
        const int c = lane + 32 * j;
        topk_insert(lv[rI], li[rI], rowp[c], cb + c);
      }
    }
    __syncthreads();
  }

  const int b = bh >> 3, head = bh & 7;

#pragma unroll
  for (int rI = 0; rI < 2; ++rI) {
    const int qrow = q0 + r0 + rI;

    // Merge 32 per-lane top-10 lists -> global top-10 (sorted desc, stable).
    float wv[TOPK];
    int   wi[TOPK];
    int cur = 0;
#pragma unroll
    for (int t = 0; t < TOPK; ++t) {
      float v = (cur < TOPK) ? lv[rI][cur] : NEG_BIG;
      int   i = (cur < TOPK) ? li[rI][cur] : 0x7fffffff;
      wave_argmax(v, i);
      wv[t] = v;
      wi[t] = i;
      if (cur < TOPK && li[rI][cur] == i) ++cur;
    }

    // Softmax over the 10 survivors; all other entries underflow to 0.0f
    // exactly (exp(-1e9 - max) == 0 in f32), matching the reference.
    const float wmax = wv[0];
    float e[TOPK], sum = 0.f;
#pragma unroll
    for (int t = 0; t < TOPK; ++t) { e[t] = expf(wv[t] - wmax); sum += e[t]; }
    const float inv = 1.0f / sum;

    // Context row: 64 dims over 32 lanes (2 each), 10 scalar FMAs per dim.
    float c0 = 0.f, c1 = 0.f;
#pragma unroll
    for (int t = 0; t < TOPK; ++t) {
      const float p = e[t] * inv;
      const float* vp = Vh + (size_t)wi[t] * DHEAD;
      c0 += p * vp[lane];
      c1 += p * vp[lane + 32];
    }
    float* ctx = out + (((size_t)b * SEQ + qrow) * DMODEL) + head * DHEAD;
    ctx[lane]      = c0;
    ctx[lane + 32] = c1;

    // Attn row: 2048 zeros via nontemporal float4 streams, then overwrite
    // the 10 probabilities (VMEM stores are in-order with stores per wave).
    float* arow = out + CTX_ELEMS + ((size_t)bh * SEQ + qrow) * SEQ;
    const f32x4 z4 = (f32x4){0.f, 0.f, 0.f, 0.f};
#pragma unroll
    for (int jj = 0; jj < 16; ++jj) {
      __builtin_nontemporal_store(z4, (f32x4*)arow + lane + 32 * jj);
    }
#pragma unroll
    for (int t = 0; t < TOPK; ++t) {
      if (lane == t) arow[wi[t]] = e[t] * inv;
    }
  }
}

// ------------------------------- launcher ----------------------------------
extern "C" void kernel_launch(void* const* d_in, const int* in_sizes, int n_in,
                              void* d_out, int out_size, void* d_ws,
                              size_t ws_size, hipStream_t stream) {
  (void)in_sizes; (void)n_in; (void)out_size; (void)ws_size;
  const float* x  = (const float*)d_in[0];
  const float* Wq = (const float*)d_in[1];
  const float* bq = (const float*)d_in[2];
  const float* Wk = (const float*)d_in[3];
  const float* bk = (const float*)d_in[4];
  const float* Wv = (const float*)d_in[5];
  const float* bv = (const float*)d_in[6];
  float* out = (float*)d_out;

  // Workspace layout (all offsets 512KB-aligned):
  char* ws = (char*)d_ws;
  __bf16* xbf = (__bf16*)(ws);                       //  8,388,608 B
  __bf16* wqb = (__bf16*)(ws + 8388608);             //    524,288 B
  __bf16* wkb = (__bf16*)(ws + 8912896);             //    524,288 B
  __bf16* wvb = (__bf16*)(ws + 9437184);             //    524,288 B
  __bf16* Qbf = (__bf16*)(ws + 9961472);             //  8,388,608 B
  __bf16* Kbf = (__bf16*)(ws + 18350080);            //  8,388,608 B
  float*  Vf  = (float*)(ws + 26738688);             // 16,777,216 B

  f32_to_bf16_kernel<<<2048, 256, 0, stream>>>(x, xbf, BATCH * SEQ * DMODEL);
  f32_to_bf16_kernel<<<256, 256, 0, stream>>>(Wq, wqb, DMODEL * DMODEL);
  f32_to_bf16_kernel<<<256, 256, 0, stream>>>(Wk, wkb, DMODEL * DMODEL);
  f32_to_bf16_kernel<<<256, 256, 0, stream>>>(Wv, wvb, DMODEL * DMODEL);

  const dim3 ggrid(64, 8);  // (M/16/8, N/64)
  proj_gemm_kernel<true ><<<ggrid, 256, 0, stream>>>(xbf, wqb, bq, (void*)Qbf);
  proj_gemm_kernel<true ><<<ggrid, 256, 0, stream>>>(xbf, wkb, bk, (void*)Kbf);
  proj_gemm_kernel<false><<<ggrid, 256, 0, stream>>>(xbf, wvb, bv, (void*)Vf);

  const dim3 agrid(SEQ / 16, BATCH * HEADS);  // (128, 32)
  sparse_attn_kernel<<<agrid, 256, 16 * 512 * sizeof(float), stream>>>(
      Qbf, Kbf, Vf, out);
}
